// GraspSampler_19078244729436
// MI455X (gfx1250) — matrix operand order repaired and checked by reading, not compiled, not run on hardware
//
#include <hip/hip_runtime.h>
#include <hip/hip_bf16.h>

typedef __attribute__((ext_vector_type(16))) _Float16     v16h;
typedef __attribute__((ext_vector_type(8)))  float        v8f;
typedef __attribute__((ext_vector_type(8)))  unsigned int v8u;

__device__ __forceinline__ unsigned int pack2(float a, float b) {
    unsigned short ua = __builtin_bit_cast(unsigned short, (_Float16)a);
    unsigned short ub = __builtin_bit_cast(unsigned short, (_Float16)b);
    return (unsigned int)ua | ((unsigned int)ub << 16);
}
__device__ __forceinline__ float lo16(unsigned int u) {
    return (float)__builtin_bit_cast(_Float16, (unsigned short)(u & 0xffffu));
}
__device__ __forceinline__ float hi16(unsigned int u) {
    return (float)__builtin_bit_cast(_Float16, (unsigned short)(u >> 16));
}

// ---------------------------------------------------------------------------
// Farthest point sampling: one block per batch element, LDS-resident dist[].
// idx[0]=0; argmax picks the FIRST max (index tie-break) like jnp.argmax.
// ---------------------------------------------------------------------------
__global__ __launch_bounds__(256) void fps_kernel(const float* __restrict__ xyz, int N, int npoint,
                                                  int* __restrict__ idxOut, float* __restrict__ newXyz)
{
    __shared__ float dist[4096];
    __shared__ float redV[256];
    __shared__ int   redI[256];
    const int b = blockIdx.x, t = threadIdx.x, T = blockDim.x;
    const float* px = xyz + (size_t)b * N * 3;
    for (int n = t; n < N; n += T) dist[n] = 1e10f;
    __syncthreads();
    int last = 0;
    for (int i = 0; i < npoint; ++i) {
        const float lx = px[last*3+0], ly = px[last*3+1], lz = px[last*3+2];
        if (t == 0) {
            idxOut[b*npoint + i] = last;
            newXyz[(size_t)(b*npoint + i)*3 + 0] = lx;
            newXyz[(size_t)(b*npoint + i)*3 + 1] = ly;
            newXyz[(size_t)(b*npoint + i)*3 + 2] = lz;
        }
        float best = -1.0f; int bi = 0;
        for (int n = t; n < N; n += T) {
            float dx = px[n*3+0]-lx, dy = px[n*3+1]-ly, dz = px[n*3+2]-lz;
            float d  = dx*dx + dy*dy + dz*dz;
            float dn = fminf(dist[n], d);
            dist[n] = dn;
            if (dn > best) { best = dn; bi = n; }
        }
        redV[t] = best; redI[t] = bi;
        __syncthreads();
        for (int s = T >> 1; s > 0; s >>= 1) {
            if (t < s) {
                if (redV[t+s] > redV[t] || (redV[t+s] == redV[t] && redI[t+s] < redI[t])) {
                    redV[t] = redV[t+s]; redI[t] = redI[t+s];
                }
            }
            __syncthreads();
        }
        last = redI[0];
        __syncthreads();
    }
}

// ---------------------------------------------------------------------------
// Ball query: thread per center; sequential scan == reference sorted-index
// semantics; pad with first hit; N-1 fallback.
// ---------------------------------------------------------------------------
__global__ void ball_query_kernel(const float* __restrict__ xyz, const float* __restrict__ newXyz,
                                  int* __restrict__ idxOut, int B, int N, int S, int nsample, float r2)
{
    const int gid = blockIdx.x * blockDim.x + threadIdx.x;
    if (gid >= B * S) return;
    const int b = gid / S;
    const float* px = xyz + (size_t)b * N * 3;
    const float* c  = newXyz + (size_t)gid * 3;
    int* out = idxOut + (size_t)gid * nsample;
    const float cx = c[0], cy = c[1], cz = c[2];
    int cnt = 0, first = N - 1;
    for (int n = 0; n < N && cnt < nsample; ++n) {
        const float dx = px[n*3+0]-cx, dy = px[n*3+1]-cy, dz = px[n*3+2]-cz;
        if (dx*dx + dy*dy + dz*dz < r2) { if (cnt == 0) first = n; out[cnt++] = n; }
    }
    for (; cnt < nsample; ++cnt) out[cnt] = first;
}

// ---------------------------------------------------------------------------
// Grouping kernels -> pair-interleaved f16 activations: u32 array [CinPad/2][M],
// word = (f16 ch2i | f16 ch2i+1 << 16). Pad channels written as zero.
// ---------------------------------------------------------------------------
__global__ void group1_kernel(const float* __restrict__ xyz, const float* __restrict__ z,
                              const float* __restrict__ newXyz, const int* __restrict__ ballIdx,
                              unsigned int* __restrict__ g, int B, int N, int S, int K, int pairs)
{
    const size_t Mtot = (size_t)B * S * K;
    const size_t m = (size_t)blockIdx.x * blockDim.x + threadIdx.x;
    if (m >= Mtot) return;
    const int s = (int)((m / K) % S);
    const int b = (int)(m / ((size_t)S * K));
    const int idx = ballIdx[m];
    const float* p = xyz    + ((size_t)b * N + idx) * 3;
    const float* c = newXyz + ((size_t)b * S + s) * 3;
    const float p0 = p[0], p1 = p[1], p2 = p[2];
    g[0*Mtot + m] = pack2(p0 - c[0], p1 - c[1]);
    g[1*Mtot + m] = pack2(p2 - c[2], p0);
    g[2*Mtot + m] = pack2(p1, p2);
    g[3*Mtot + m] = pack2(z[b*2+0], z[b*2+1]);
    for (int i = 4; i < pairs; ++i) g[(size_t)i*Mtot + m] = 0u;
}

__global__ void group2_kernel(const float* __restrict__ srcXyz,   // [B][S1][3]
                              const float* __restrict__ newXyz,   // [B][S][3]
                              const int* __restrict__ ballIdx,
                              const _Float16* __restrict__ pooled, // [C][B*S1]
                              unsigned int* __restrict__ g,
                              int B, int S1, int S, int K, int C, int pairs)
{
    const size_t Mtot = (size_t)B * S * K;
    const size_t m = (size_t)blockIdx.x * blockDim.x + threadIdx.x;
    if (m >= Mtot) return;
    const int s = (int)((m / K) % S);
    const int b = (int)(m / ((size_t)S * K));
    const int idx = ballIdx[m];
    const float* p = srcXyz + ((size_t)b * S1 + idx) * 3;
    const float* c = newXyz + ((size_t)b * S  + s  ) * 3;
    float d[3] = { p[0]-c[0], p[1]-c[1], p[2]-c[2] };
    const size_t SB  = (size_t)B * S1;
    const size_t col = (size_t)b * S1 + idx;
    auto chan = [&](int ch) -> float {
        if (ch < 3) return d[ch];
        ch -= 3;
        return (ch < C) ? (float)pooled[(size_t)ch * SB + col] : 0.f;
    };
    for (int i = 0; i < pairs; ++i)
        g[(size_t)i*Mtot + m] = pack2(chan(2*i), chan(2*i+1));
}

__global__ void group3_kernel(const float* __restrict__ newXyz2, const _Float16* __restrict__ pooled2,
                              unsigned int* __restrict__ g, int Mtot, int C, int pairs)
{
    const int m = blockIdx.x * blockDim.x + threadIdx.x;
    if (m >= Mtot) return;
    auto chan = [&](int ch) -> float {
        if (ch < 3) return newXyz2[m*3 + ch];
        ch -= 3;
        return (ch < C) ? (float)pooled2[(size_t)ch * Mtot + m] : 0.f;
    };
    for (int i = 0; i < pairs; ++i)
        g[(size_t)i*(size_t)Mtot + m] = pack2(chan(2*i), chan(2*i+1));
}

// ---------------------------------------------------------------------------
// Pre-pack W[Cout][Cin] f32 -> f16 A-fragments, layout [ks][ot][lane][16]
// (ot-tiles of one K-step adjacent so a wave can clause-load 4 fragments),
// zero-padded to CinPad. A-layout (wave32): K = 16*half + e (contiguous, same order as the B loads).
// ---------------------------------------------------------------------------
__global__ void pack_w_kernel(const float* __restrict__ W, _Float16* __restrict__ Wp,
                              int Cout, int Cin, int CinPad)
{
    const int total = Cout * CinPad;
    const int i = blockIdx.x * blockDim.x + threadIdx.x;
    if (i >= total) return;
    const int oTiles = Cout >> 4;
    const int e    = i & 15;
    const int lane = (i >> 4) & 31;
    const int rest = i >> 9;
    const int ot   = rest % oTiles;
    const int ks   = rest / oTiles;
    const int half = lane >> 4, row = lane & 15;
    const int v = e >> 1, p = e & 1;
    const int k = ks*32 + half*16 + e;
    Wp[i] = (k < Cin) ? (_Float16)W[(size_t)(ot*16 + row)*Cin + k] : (_Float16)0.f;
}

// ---------------------------------------------------------------------------
// WMMA GEMM, templated on (CinPad, M, Cout, BIASED); all strides compile-time.
// Each wave: OT=4 Cout-tiles x one M-tile. Per K-step one B fragment feeds 4
// independent v_wmma_f32_16x16x32_f16. The K loop is software-pipelined: A
// loads + NEXT B loads issue before consuming the current B, so (in-order VMEM
// retirement) the WMMAs can proceed on a partial loadcnt wait while next-B is
// in flight. Y stores are non-temporal (268MB streaming buffer > 192MB L2).
// ---------------------------------------------------------------------------
template<int CPAD, int MM, int COUT, bool BIASED>
__global__ __launch_bounds__(256) void wmma_gemm_kernel(const _Float16* __restrict__ Wp,
                                                        const unsigned int* __restrict__ X,
                                                        float* __restrict__ Y,
                                                        const float* __restrict__ bias)
{
    constexpr int OT      = 4;
    constexpr int kSteps  = CPAD / 32;
    constexpr int oTiles  = COUT / 16;
    constexpr int oGroups = oTiles / OT;
    constexpr int total   = oGroups * (MM / 16);
    const int wave = blockIdx.x * (blockDim.x >> 5) + ((int)threadIdx.x >> 5);
    if (wave >= total) return;
    const int mt = wave / oGroups, og = wave - mt * oGroups;
    const int M0 = mt << 4;
    const int lane = threadIdx.x & 31;
    const int nrow = lane & 15;     // A row / B col (pixel) / D col
    const int half = lane >> 4;

    const _Float16*     wpBase = Wp + ((size_t)og * OT * 32 + lane) * 16;
    const unsigned int* xc     = X + (size_t)(half * 8) * MM + M0 + nrow;

    auto loadB = [&](int ks) -> v8u {
        v8u bu;
#pragma unroll
        for (int j = 0; j < 8; ++j)
            bu[j] = xc[(size_t)(ks*16 + j) * MM];
        return bu;
    };

    const v8f vzero = {0.f,0.f,0.f,0.f,0.f,0.f,0.f,0.f};
    v8f acc[OT];
#pragma unroll
    for (int i = 0; i < OT; ++i) acc[i] = vzero;

    v8u bcur = loadB(0);
#pragma unroll
    for (int ks = 0; ks < kSteps; ++ks) {
        const _Float16* wk = wpBase + (size_t)ks * oTiles * 512;
        const v16h a0 = *(const v16h*)(wk + 0*512);    // 32B: 2x b128 each
        const v16h a1 = *(const v16h*)(wk + 1*512);
        const v16h a2 = *(const v16h*)(wk + 2*512);
        const v16h a3 = *(const v16h*)(wk + 3*512);
        v8u bnext = {};
        if (ks + 1 < kSteps) bnext = loadB(ks + 1);    // prefetch next B
        const v16h b = __builtin_bit_cast(v16h, bcur);
        acc[0] = __builtin_amdgcn_wmma_f32_16x16x32_f16(false, a0, false, b, (short)0, acc[0], false, false);
        acc[1] = __builtin_amdgcn_wmma_f32_16x16x32_f16(false, a1, false, b, (short)0, acc[1], false, false);
        acc[2] = __builtin_amdgcn_wmma_f32_16x16x32_f16(false, a2, false, b, (short)0, acc[2], false, false);
        acc[3] = __builtin_amdgcn_wmma_f32_16x16x32_f16(false, a3, false, b, (short)0, acc[3], false, false);
        bcur = bnext;
    }
#pragma unroll
    for (int i = 0; i < OT; ++i) {
        const int O = (og * OT + i) << 4;
#pragma unroll
        for (int v = 0; v < 8; ++v) {
            const int row = O + v + half * 8;   // D: m = v + 8*half, n = lane&15
            float val = acc[i][v];
            if constexpr (BIASED) val += bias[row];
            __builtin_nontemporal_store(val, &Y[(size_t)row * MM + M0 + nrow]);
        }
    }
}

template<int CPAD, int MM, int COUT, bool BIASED>
static inline void launch_gemm(const _Float16* Wp, const unsigned int* X, float* Y,
                               const float* bias, hipStream_t stream)
{
    constexpr int tiles = (COUT / 64) * (MM / 16);   // oGroups * mTiles
    wmma_gemm_kernel<CPAD, MM, COUT, BIASED><<<(tiles + 7)/8, 256, 0, stream>>>(Wp, X, Y, bias);
}

// ---------------------------------------------------------------------------
// Training-mode BN (+ReLU): one block per CHANNEL PAIR; f32 stats (biased var,
// eps=1e-5) on the GEMM output, writes packed-f16 next-layer input (coalesced).
// ---------------------------------------------------------------------------
__global__ __launch_bounds__(256) void bn_relu_pack_kernel(const float* __restrict__ Y,
                                                           const float* __restrict__ gamma,
                                                           const float* __restrict__ beta,
                                                           unsigned int* __restrict__ Xout, int M)
{
    const int cp = blockIdx.x, t = threadIdx.x, T = blockDim.x;
    const float* r0 = Y + (size_t)(2*cp) * M;
    const float* r1 = r0 + M;
    __shared__ float sA[256], sB[256], sC[256], sD[256];
    float a0 = 0.f, q0 = 0.f, a1 = 0.f, q1 = 0.f;
    for (int m = t; m < M; m += T) {
        const float v0 = r0[m], v1 = r1[m];
        a0 += v0; q0 += v0*v0; a1 += v1; q1 += v1*v1;
    }
    sA[t] = a0; sB[t] = q0; sC[t] = a1; sD[t] = q1;
    __syncthreads();
    for (int s = T >> 1; s > 0; s >>= 1) {
        if (t < s) { sA[t]+=sA[t+s]; sB[t]+=sB[t+s]; sC[t]+=sC[t+s]; sD[t]+=sD[t+s]; }
        __syncthreads();
    }
    const float inv = 1.f / (float)M;
    const float mu0 = sA[0]*inv, mu1 = sC[0]*inv;
    const float g0 = gamma[2*cp+0] * rsqrtf(fmaxf(sB[0]*inv - mu0*mu0, 0.f) + 1e-5f);
    const float g1 = gamma[2*cp+1] * rsqrtf(fmaxf(sD[0]*inv - mu1*mu1, 0.f) + 1e-5f);
    const float b0 = beta[2*cp+0], b1 = beta[2*cp+1];
    unsigned int* out = Xout + (size_t)cp * M;
    for (int m = t; m < M; m += T) {
        const float v0 = fmaxf((r0[m]-mu0)*g0 + b0, 0.f);
        const float v1 = fmaxf((r1[m]-mu1)*g1 + b1, 0.f);
        out[m] = pack2(v0, v1);
    }
}

// Max over K from packed pairs; plain f16 output (for gather) ...
__global__ void maxpool_f16_kernel(const unsigned int* __restrict__ Xp, _Float16* __restrict__ out,
                                   int SB, int K, int totalPairs)
{
    const int i = blockIdx.x * blockDim.x + threadIdx.x;
    if (i >= totalPairs) return;
    const int cp = i / SB, sb = i - cp * SB;
    const unsigned int* row = Xp + ((size_t)cp * SB + sb) * K;
    unsigned int u = row[0];
    float m0 = lo16(u), m1 = hi16(u);
    for (int k = 1; k < K; ++k) { u = row[k]; m0 = fmaxf(m0, lo16(u)); m1 = fmaxf(m1, hi16(u)); }
    out[(size_t)(2*cp+0) * SB + sb] = (_Float16)m0;
    out[(size_t)(2*cp+1) * SB + sb] = (_Float16)m1;
}

// ... or packed-pair output (direct GEMM input for the FC stage).
__global__ void maxpool_pack_kernel(const unsigned int* __restrict__ Xp, unsigned int* __restrict__ out,
                                    int SB, int K, int totalPairs)
{
    const int i = blockIdx.x * blockDim.x + threadIdx.x;
    if (i >= totalPairs) return;
    const int cp = i / SB, sb = i - cp * SB;
    const unsigned int* row = Xp + ((size_t)cp * SB + sb) * K;
    unsigned int u = row[0];
    float m0 = lo16(u), m1 = hi16(u);
    for (int k = 1; k < K; ++k) { u = row[k]; m0 = fmaxf(m0, lo16(u)); m1 = fmaxf(m1, hi16(u)); }
    out[(size_t)cp * SB + sb] = pack2(m0, m1);
}

// ---------------------------------------------------------------------------
// Heads: block per batch row; X = packed pairs [512][B] of the 1024-ch FC out.
// q L2-normalized (1e-12 floor), t, sigmoid(conf). d_out = qt[64][7] ++ conf[64].
// ---------------------------------------------------------------------------
__global__ __launch_bounds__(256) void heads_kernel(const unsigned int* __restrict__ Xp,
                                                    const float* __restrict__ qW, const float* __restrict__ qb,
                                                    const float* __restrict__ tW, const float* __restrict__ tb,
                                                    const float* __restrict__ cW, const float* __restrict__ cb,
                                                    float* __restrict__ out, int B)
{
    const int b = blockIdx.x, t = threadIdx.x, T = blockDim.x;
    float acc[8];
#pragma unroll
    for (int o = 0; o < 8; ++o) acc[o] = 0.f;
    for (int cp = t; cp < 512; cp += T) {
        const unsigned int u = Xp[(size_t)cp * B + b];
        const float x0 = lo16(u), x1 = hi16(u);
        const int c = 2 * cp;
#pragma unroll
        for (int o = 0; o < 4; ++o) acc[o]   += qW[o*1024 + c]*x0 + qW[o*1024 + c + 1]*x1;
#pragma unroll
        for (int o = 0; o < 3; ++o) acc[4+o] += tW[o*1024 + c]*x0 + tW[o*1024 + c + 1]*x1;
        acc[7] += cW[c]*x0 + cW[c+1]*x1;
    }
    __shared__ float red[8][256];
#pragma unroll
    for (int o = 0; o < 8; ++o) red[o][t] = acc[o];
    __syncthreads();
    for (int s = T >> 1; s > 0; s >>= 1) {
        if (t < s) {
#pragma unroll
            for (int o = 0; o < 8; ++o) red[o][t] += red[o][t+s];
        }
        __syncthreads();
    }
    if (t == 0) {
        float q[4], n2 = 0.f;
#pragma unroll
        for (int o = 0; o < 4; ++o) { q[o] = red[o][0] + qb[o]; n2 += q[o]*q[o]; }
        const float nrm = fmaxf(sqrtf(n2), 1e-12f);
#pragma unroll
        for (int o = 0; o < 4; ++o) out[b*7 + o] = q[o] / nrm;
#pragma unroll
        for (int j = 0; j < 3; ++j) out[b*7 + 4 + j] = red[4+j][0] + tb[j];
        out[B*7 + b] = 1.f / (1.f + expf(-(red[7][0] + cb[0])));
    }
}

// ---------------------------------------------------------------------------
extern "C" void kernel_launch(void* const* d_in, const int* in_sizes, int n_in,
                              void* d_out, int out_size, void* d_ws, size_t ws_size,
                              hipStream_t stream)
{
    (void)in_sizes; (void)n_in; (void)out_size; (void)ws_size;
    const float* xyz = (const float*)d_in[0];
    const float* z   = (const float*)d_in[1];
    // params flattened in jax pytree (sorted-key) order after xyz, z:
    // conf{W,b}, fc[0]{W,b,beta,gamma}, fc[1]{...}, q{W,b},
    // sa[0..2][0..2]{W,beta,gamma}, t{W,b}
    const int P = 2;
    const float* confW    = (const float*)d_in[P+0];
    const float* confb    = (const float*)d_in[P+1];
    const float* fc0W     = (const float*)d_in[P+2];
    const float* fc0b     = (const float*)d_in[P+3];
    const float* fc0beta  = (const float*)d_in[P+4];
    const float* fc0gamma = (const float*)d_in[P+5];
    const float* fc1W     = (const float*)d_in[P+6];
    const float* fc1b     = (const float*)d_in[P+7];
    const float* fc1beta  = (const float*)d_in[P+8];
    const float* fc1gamma = (const float*)d_in[P+9];
    const float* qW       = (const float*)d_in[P+10];
    const float* qb       = (const float*)d_in[P+11];
    const float *saW[3][3], *saBeta[3][3], *saGamma[3][3];
    int li = P + 12;
    for (int s = 0; s < 3; ++s)
        for (int l = 0; l < 3; ++l) {
            saW[s][l]     = (const float*)d_in[li++];
            saBeta[s][l]  = (const float*)d_in[li++];
            saGamma[s][l] = (const float*)d_in[li++];
        }
    const float* tW = (const float*)d_in[li++];
    const float* tb = (const float*)d_in[li++];

    const int B = 64, N = 4096, S1 = 128, K1 = 64, S2 = 32, K2 = 128, K3 = 32;
    constexpr int M1 = 64*128*64;   // 524288
    constexpr int M2 = 64*32*128;   // 262144
    constexpr int M3 = 64*32;       // 2048
    constexpr int MB = 64;

    char* ws = (char*)d_ws;
    size_t off = 0;
    auto alloc = [&](size_t bytes) -> char* {
        char* p = ws + off;
        off = (off + bytes + 255) & ~(size_t)255;
        return p;
    };
    int*      fpsIdx1  = (int*)     alloc((size_t)B*S1*4);
    float*    newXyz1  = (float*)   alloc((size_t)B*S1*3*4);
    int*      ballIdx1 = (int*)     alloc((size_t)M1*4);
    int*      fpsIdx2  = (int*)     alloc((size_t)B*S2*4);
    float*    newXyz2  = (float*)   alloc((size_t)B*S2*3*4);
    int*      ballIdx2 = (int*)     alloc((size_t)M2*4);
    _Float16* pooled1  = (_Float16*)alloc((size_t)128*B*S1*2);
    _Float16* pooled2  = (_Float16*)alloc((size_t)256*B*S2*2);
    unsigned int* pooled3 = (unsigned int*)alloc((size_t)256*B*4);   // packed pairs [512/2][B]
    // packed weights (f16): [Cout * CinPad] each
    const int CoutA[11] = {  64,  64, 128, 128, 128, 256, 256, 256, 512, 1024, 1024 };
    const int CinA [11] = {   8,  64,  64, 131, 128, 128, 259, 256, 256,  512, 1024 };
    const int CpadA[11] = {  32,  64,  64, 160, 128, 128, 288, 256, 256,  512, 1024 };
    _Float16* wp[11];
    for (int i = 0; i < 11; ++i) wp[i] = (_Float16*)alloc((size_t)CoutA[i]*CpadA[i]*2);
    // activations: pair-interleaved u32 [CinPad/2][M]; max = 128ch x M1 -> 64*M1 words
    unsigned int* Xa = (unsigned int*)alloc((size_t)64*M1*4);
    unsigned int* Xb = (unsigned int*)alloc((size_t)64*M1*4);
    float*        Yf = (float*)      alloc((size_t)128*M1*4);        // f32 GEMM out (max 128 x M1)
    (void)fpsIdx1; (void)fpsIdx2;

    auto packw = [&](int i, const float* W) {
        const int total = CoutA[i]*CpadA[i];
        pack_w_kernel<<<(total+255)/256, 256, 0, stream>>>(W, wp[i], CoutA[i], CinA[i], CpadA[i]);
    };
    auto bnp = [&](const float* ga, const float* be, unsigned int* Xout, int C, int M) {
        bn_relu_pack_kernel<<<C/2, 256, 0, stream>>>(Yf, ga, be, Xout, M);
    };

    // weight packing (cheap, ~3.9 MB total)
    packw(0, saW[0][0]); packw(1, saW[0][1]); packw(2, saW[0][2]);
    packw(3, saW[1][0]); packw(4, saW[1][1]); packw(5, saW[1][2]);
    packw(6, saW[2][0]); packw(7, saW[2][1]); packw(8, saW[2][2]);
    packw(9, fc0W);      packw(10, fc1W);

    // ---------------- SA1: npoint=128, r=0.04, nsample=64, mlp 8->64->64->128
    fps_kernel<<<B, 256, 0, stream>>>(xyz, N, S1, fpsIdx1, newXyz1);
    ball_query_kernel<<<(B*S1+255)/256, 256, 0, stream>>>(xyz, newXyz1, ballIdx1, B, N, S1, K1, 0.0016f);
    group1_kernel<<<(M1+255)/256, 256, 0, stream>>>(xyz, z, newXyz1, ballIdx1, Xa, B, N, S1, K1, 16);
    launch_gemm< 32, M1,  64, false>(wp[0], Xa, Yf, nullptr, stream); bnp(saGamma[0][0], saBeta[0][0], Xb,  64, M1);
    launch_gemm< 64, M1,  64, false>(wp[1], Xb, Yf, nullptr, stream); bnp(saGamma[0][1], saBeta[0][1], Xa,  64, M1);
    launch_gemm< 64, M1, 128, false>(wp[2], Xa, Yf, nullptr, stream); bnp(saGamma[0][2], saBeta[0][2], Xb, 128, M1);
    maxpool_f16_kernel<<<(64*B*S1+255)/256, 256, 0, stream>>>(Xb, pooled1, B*S1, K1, 64*B*S1);

    // ---------------- SA2: npoint=32, r=0.04, nsample=128, mlp 131->128->128->256
    fps_kernel<<<B, 256, 0, stream>>>(newXyz1, S1, S2, fpsIdx2, newXyz2);
    ball_query_kernel<<<(B*S2+255)/256, 256, 0, stream>>>(newXyz1, newXyz2, ballIdx2, B, S1, S2, K2, 0.0016f);
    group2_kernel<<<(M2+255)/256, 256, 0, stream>>>(newXyz1, newXyz2, ballIdx2, pooled1, Xa, B, S1, S2, K2, 128, 80);
    launch_gemm<160, M2, 128, false>(wp[3], Xa, Yf, nullptr, stream); bnp(saGamma[1][0], saBeta[1][0], Xb, 128, M2);
    launch_gemm<128, M2, 128, false>(wp[4], Xb, Yf, nullptr, stream); bnp(saGamma[1][1], saBeta[1][1], Xa, 128, M2);
    launch_gemm<128, M2, 256, false>(wp[5], Xa, Yf, nullptr, stream); bnp(saGamma[1][2], saBeta[1][2], Xb, 256, M2);
    maxpool_f16_kernel<<<(128*B*S2+255)/256, 256, 0, stream>>>(Xb, pooled2, B*S2, K2, 128*B*S2);

    // ---------------- SA3: global group, mlp 259->256->256->512
    group3_kernel<<<(M3+255)/256, 256, 0, stream>>>(newXyz2, pooled2, Xa, M3, 256, 144);
    launch_gemm<288, M3, 256, false>(wp[6], Xa, Yf, nullptr, stream); bnp(saGamma[2][0], saBeta[2][0], Xb, 256, M3);
    launch_gemm<256, M3, 256, false>(wp[7], Xb, Yf, nullptr, stream); bnp(saGamma[2][1], saBeta[2][1], Xa, 256, M3);
    launch_gemm<256, M3, 512, false>(wp[8], Xa, Yf, nullptr, stream); bnp(saGamma[2][2], saBeta[2][2], Xb, 512, M3);
    maxpool_pack_kernel<<<(256*B+255)/256, 256, 0, stream>>>(Xb, pooled3, B, K3, 256*B);

    // ---------------- FC 512->1024->1024 (+bias, BN over batch, ReLU), heads
    launch_gemm< 512, MB, 1024, true>(wp[9],  pooled3, Yf, fc0b, stream); bnp(fc0gamma, fc0beta, Xa, 1024, MB);
    launch_gemm<1024, MB, 1024, true>(wp[10], Xa,      Yf, fc1b, stream); bnp(fc1gamma, fc1beta, Xb, 1024, MB);
    heads_kernel<<<B, 256, 0, stream>>>(Xb, qW, qb, tW, tb, confW, confb, (float*)d_out, B);
}
